// SceneEncoder_18915035971760
// MI455X (gfx1250) — matrix operand order repaired and checked by reading, not compiled, not run on hardware
//
#include <hip/hip_runtime.h>
#include <hip/hip_bf16.h>

// ---------------- types / helpers ----------------
typedef __attribute__((ext_vector_type(16))) __bf16 bf16x16;
typedef __attribute__((ext_vector_type(8)))  __bf16 bf16x8;
typedef __attribute__((ext_vector_type(8)))  float  f32x8;

__device__ __forceinline__ unsigned short f2bf(float f) {
    unsigned int u = __float_as_uint(f);
    unsigned int r = u + 0x7FFFu + ((u >> 16) & 1u);   // round-to-nearest-even
    return (unsigned short)(r >> 16);
}
__device__ __forceinline__ float bf2f(unsigned short h) {
    return __uint_as_float(((unsigned int)h) << 16);
}

#if defined(__has_builtin)
#if __has_builtin(__builtin_amdgcn_cvt_pk_bf16_f32)
#define HAVE_PK_BF16 1
#endif
#if __has_builtin(__builtin_amdgcn_global_load_async_to_lds_b128) && \
    __has_builtin(__builtin_amdgcn_s_wait_asynccnt)
#define HAVE_ASYNC_LDS 1
#endif
#endif

// pack two f32 into one dword of bf16 (RNE): 2x(bfe+add3) + v_perm_b32
__device__ __forceinline__ unsigned int pack2(float a, float b) {
#ifdef HAVE_PK_BF16
    auto t = __builtin_amdgcn_cvt_pk_bf16_f32(a, b);
    unsigned int r; __builtin_memcpy(&r, &t, 4); return r;
#else
    unsigned int ua = __float_as_uint(a), ub = __float_as_uint(b);
    ua += 0x7FFFu + ((ua >> 16) & 1u);
    ub += 0x7FFFu + ((ub >> 16) & 1u);
    // result = [ub.hi16 : ua.hi16]; src0=ub supplies bytes 4..7, src1=ua bytes 0..3
    return __builtin_amdgcn_perm(ub, ua, 0x07060302u);
#endif
}
__device__ __forceinline__ bf16x16 cat8(bf16x8 lo, bf16x8 hi) {
    return __builtin_shufflevector(lo, hi, 0,1,2,3,4,5,6,7,8,9,10,11,12,13,14,15);
}

#ifdef HAVE_ASYNC_LDS
// Builtin signature (from hipcc diagnostic): param0 is
//   'int __attribute__((vector_size(16))) __device__ *'  (addrspace(1) v4i*)
typedef int v4i_t __attribute__((vector_size(16)));
typedef __attribute__((address_space(1))) v4i_t* as1_v4i;
typedef __attribute__((address_space(3))) v4i_t* as3_v4i;
__device__ __forceinline__ void async_ld_b128(const void* g, void* l) {
    __builtin_amdgcn_global_load_async_to_lds_b128((as1_v4i)g, (as3_v4i)l, 0, 0);
}
#endif

// ---------------- WMMA GEMM: C[M,N] = alpha * A[M,K] * B[N,K]^T (+bias) ----------------
// A: bf16 row-major (lda). B: f32 (converted inline via v_perm pack) or bf16, row-major N x K.
// Double-buffered LDS, software-pipelined; A (and bf16 B) staged via
// GLOBAL_LOAD_ASYNC_TO_LDS_B128 when available. Batched via blockIdx.z.
#define BM 128
#define BN 64
#define BKK 32
#define LDSP (BKK + 8)

template<bool B_F32, bool OUT_F32, bool RELU>
__global__ __launch_bounds__(256) void gemm_bt(
    const unsigned short* __restrict__ Ag,
    const void* __restrict__ Bg,
    void* __restrict__ Cg,
    int K, int lda, int ldb, int ldc,
    long sA, long sB, long sC,
    float alpha, const float* __restrict__ bias)
{
    __shared__ __align__(16) unsigned short As[2][BM][LDSP];
    __shared__ __align__(16) unsigned short Bs[2][BN][LDSP];

    const int tid  = threadIdx.x;
    const int wave = tid >> 5;
    const int lane = tid & 31;
    const int hw   = lane >> 4;     // half-wave select
    const int lr   = lane & 15;

    const int bm = blockIdx.y * BM;
    const int bn = blockIdx.x * BN;
    const long z = blockIdx.z;

    const unsigned short* A  = Ag + z * sA;
    const float*          Bf = (const float*)Bg + z * sB;
    const unsigned short* Bh = (const unsigned short*)Bg + z * sB;

    const int wmi = (wave >> 1) * 32;  // 4 waves in M
    const int wni = (wave & 1) * 32;   // 2 waves in N

    f32x8 acc[2][2];
    const f32x8 zero = {0.f,0.f,0.f,0.f,0.f,0.f,0.f,0.f};
    #pragma unroll
    for (int i = 0; i < 2; ++i)
        #pragma unroll
        for (int j = 0; j < 2; ++j) acc[i][j] = zero;

    // cooperative-load mappings
    const int arow = tid >> 1;          // 0..127
    const int acol = (tid & 1) * 16;    // 0 / 16
    const int brow = tid >> 2;          // 0..63
    const int bcol = (tid & 3) * 8;     // 0,8,16,24

    const unsigned short* aSrcBase = A + (long)(bm + arow) * lda + acol;
    const float*          bSrcF    = Bf + (long)(bn + brow) * ldb + bcol;
    const unsigned short* bSrcH    = Bh + (long)(bn + brow) * ldb + bcol;

    uint4 ra0, ra1, rb;

    // ---- prologue: stage tile 0 into buffer 0 ----
#ifdef HAVE_ASYNC_LDS
    async_ld_b128(aSrcBase,     &As[0][arow][acol]);
    async_ld_b128(aSrcBase + 8, &As[0][arow][acol + 8]);
    if (B_F32) {
        const float4* b4 = (const float4*)bSrcF;
        float4 q0 = b4[0], q1 = b4[1];
        rb = make_uint4(pack2(q0.x,q0.y), pack2(q0.z,q0.w),
                        pack2(q1.x,q1.y), pack2(q1.z,q1.w));
        *(uint4*)&Bs[0][brow][bcol] = rb;
    } else {
        async_ld_b128(bSrcH, &Bs[0][brow][bcol]);
    }
    __builtin_amdgcn_s_wait_asynccnt(0);
    __syncthreads();
#else
    {
        const uint4* s4 = (const uint4*)aSrcBase;
        ra0 = s4[0]; ra1 = s4[1];
        if (B_F32) {
            const float4* b4 = (const float4*)bSrcF;
            float4 q0 = b4[0], q1 = b4[1];
            rb = make_uint4(pack2(q0.x,q0.y), pack2(q0.z,q0.w),
                            pack2(q1.x,q1.y), pack2(q1.z,q1.w));
        } else {
            rb = *(const uint4*)bSrcH;
        }
    }
    *(uint4*)&As[0][arow][acol]     = ra0;
    *(uint4*)&As[0][arow][acol + 8] = ra1;
    *(uint4*)&Bs[0][brow][bcol]     = rb;
    __syncthreads();
#endif

    int p = 0;
    for (int k0 = 0; k0 < K; k0 += BKK) {
        // ---- fetch next tile (clamped reload on last iter) ----
        const int kn = (k0 + BKK < K) ? (k0 + BKK) : k0;
#ifdef HAVE_ASYNC_LDS
        // direct global -> LDS (next buffer); buffer p^1 was last read before the
        // previous barrier, so writes here cannot race reads.
        async_ld_b128(aSrcBase + kn,     &As[p ^ 1][arow][acol]);
        async_ld_b128(aSrcBase + kn + 8, &As[p ^ 1][arow][acol + 8]);
        if (B_F32) {
            const float* src = bSrcF + kn;
            const float4* b4 = (const float4*)src;
            float4 q0 = b4[0], q1 = b4[1];
            rb = make_uint4(pack2(q0.x,q0.y), pack2(q0.z,q0.w),
                            pack2(q1.x,q1.y), pack2(q1.z,q1.w));
            __builtin_prefetch(src + BKK, 0, 1);
        } else {
            async_ld_b128(bSrcH + kn, &Bs[p ^ 1][brow][bcol]);
        }
#else
        {
            const uint4* s4 = (const uint4*)(aSrcBase + kn);
            ra0 = s4[0]; ra1 = s4[1];
        }
        if (B_F32) {
            const float* src = bSrcF + kn;
            const float4* b4 = (const float4*)src;
            float4 q0 = b4[0], q1 = b4[1];
            rb = make_uint4(pack2(q0.x,q0.y), pack2(q0.z,q0.w),
                            pack2(q1.x,q1.y), pack2(q1.z,q1.w));
            __builtin_prefetch(src + BKK, 0, 1);
        } else {
            const unsigned short* src = bSrcH + kn;
            rb = *(const uint4*)src;
            __builtin_prefetch(src + BKK, 0, 1);
        }
#endif

        // ---- compute from buffer p ----
        // A frag 16x32 bf16: lanes 0-15 row m K{0-7,16-23}; lanes 16-31 K{8-15,24-31}
        bf16x16 af[2];
        #pragma unroll
        for (int mt = 0; mt < 2; ++mt) {
            const int r = wmi + mt * 16 + lr;
            bf16x8 lo = *(const bf16x8*)&As[p][r][hw * 8];
            bf16x8 hi = *(const bf16x8*)&As[p][r][16 + hw * 8];
            af[mt] = cat8(lo, hi);
        }
        // B frag 32x16 bf16: lane n = lr; lanes 0-15 K{0-7,16-23}; lanes 16-31 K{8-15,24-31}
        bf16x16 bfv[2];
        #pragma unroll
        for (int nt = 0; nt < 2; ++nt) {
            const int c = wni + nt * 16 + lr;
            bf16x8 lo = *(const bf16x8*)&Bs[p][c][hw * 8];
            bf16x8 hi = *(const bf16x8*)&Bs[p][c][16 + hw * 8];
            bfv[nt] = cat8(lo, hi);
        }
        #pragma unroll
        for (int mt = 0; mt < 2; ++mt)
            #pragma unroll
            for (int nt = 0; nt < 2; ++nt)
                acc[mt][nt] = __builtin_amdgcn_wmma_f32_16x16x32_bf16(
                    false, af[mt], false, bfv[nt], (short)0, acc[mt][nt], false, false);

        // ---- publish next tile ----
#ifdef HAVE_ASYNC_LDS
        if (B_F32) *(uint4*)&Bs[p ^ 1][brow][bcol] = rb;
        __builtin_amdgcn_s_wait_asynccnt(0);
#else
        *(uint4*)&As[p ^ 1][arow][acol]     = ra0;
        *(uint4*)&As[p ^ 1][arow][acol + 8] = ra1;
        *(uint4*)&Bs[p ^ 1][brow][bcol]     = rb;
#endif
        __syncthreads();
        p ^= 1;
    }

    // ---- epilogue: C layout VGPR v -> M = v + hw*8, N = lr ----
    #pragma unroll
    for (int mt = 0; mt < 2; ++mt) {
        #pragma unroll
        for (int nt = 0; nt < 2; ++nt) {
            const int cn = bn + wni + nt * 16 + lr;
            const float bv = bias ? bias[cn] : 0.0f;
            #pragma unroll
            for (int v = 0; v < 8; ++v) {
                const int cm = bm + wmi + mt * 16 + hw * 8 + v;
                float val = acc[mt][nt][v] * alpha + bv;
                if (RELU) val = fmaxf(val, 0.0f);
                if (OUT_F32)
                    ((float*)Cg + z * sC)[(long)cm * ldc + cn] = val;
                else
                    ((unsigned short*)Cg + z * sC)[(long)cm * ldc + cn] = f2bf(val);
            }
        }
    }
}

// ---------------- small VALU kernels ----------------

// x (1,A,T,F) f32 -> X0 bf16 [t][a*16+f]   (1 a t f -> t (a f))
__global__ void pack_x(const float* __restrict__ x, unsigned short* __restrict__ o) {
    int i = blockIdx.x * 256 + threadIdx.x;            // over 1M
    int f = i & 15, a = (i >> 4) & 255, t = i >> 12;
    o[i] = f2bf(x[a * 4096 + t * 16 + f]);
}

// O2 [t][a*16+f] bf16 -> H2 [a][t*16+f]    (t (a f) -> a (t f))
__global__ void transpose_af(const unsigned short* __restrict__ in,
                             unsigned short* __restrict__ out) {
    int o = blockIdx.x * 256 + threadIdx.x;            // over 1M
    int f = o & 15, t = (o >> 4) & 255, a = o >> 12;
    out[o] = in[t * 4096 + a * 16 + f];
}

// NeoX half-split RoPE in-place on Q and K (S=256 rows, d=4096, half=2048)
__global__ void rope_qk(unsigned short* __restrict__ Q, unsigned short* __restrict__ Km) {
    int idx = blockIdx.x * 256 + threadIdx.x;          // 256*2048
    int t = idx >> 11, i = idx & 2047;
    float freq = __expf(-(2.0f * (float)i / 4096.0f) * __logf(10000.0f));
    float ang = (float)t * freq;
    float c = __cosf(ang), s = __sinf(ang);
    long p0 = (long)t * 4096 + i, p1 = p0 + 2048;
    {   float x1 = bf2f(Q[p0]), x2 = bf2f(Q[p1]);
        Q[p0] = f2bf(x1 * c - x2 * s);
        Q[p1] = f2bf(x2 * c + x1 * s); }
    {   float x1 = bf2f(Km[p0]), x2 = bf2f(Km[p1]);
        Km[p0] = f2bf(x1 * c - x2 * s);
        Km[p1] = f2bf(x2 * c + x1 * s); }
}

// row-wise softmax over 256 logits (f32 in) -> bf16 probs
__global__ __launch_bounds__(256) void softmax_rows(const float* __restrict__ S,
                                                    unsigned short* __restrict__ P) {
    __shared__ float red[256];
    int row = blockIdx.x, t = threadIdx.x;
    float v = S[(long)row * 256 + t];
    red[t] = v; __syncthreads();
    #pragma unroll
    for (int s = 128; s > 0; s >>= 1) { if (t < s) red[t] = fmaxf(red[t], red[t + s]); __syncthreads(); }
    float m = red[0]; __syncthreads();
    float e = __expf(v - m);
    red[t] = e; __syncthreads();
    #pragma unroll
    for (int s = 128; s > 0; s >>= 1) { if (t < s) red[t] += red[t + s]; __syncthreads(); }
    P[(long)row * 256 + t] = f2bf(e / red[0]);
}

// O[s, h*hd + d] = sum_S P[h,s,S] * V[S, h*hd + d]  (row width fixed = 4096)
__global__ __launch_bounds__(256) void attn_av(const unsigned short* __restrict__ P,
                                               const unsigned short* __restrict__ V,
                                               unsigned short* __restrict__ O, int hd) {
    __shared__ float p[256];
    int s = blockIdx.x, h = blockIdx.y, t = threadIdx.x;
    p[t] = bf2f(P[((long)h * 256 + s) * 256 + t]);
    __syncthreads();
    int ncol = hd >> 8;                                 // 16 (time) or 2 (agent)
    for (int c = 0; c < ncol; ++c) {
        int col = h * hd + c * 256 + t;
        float sum = 0.0f;
        #pragma unroll 4
        for (int S2 = 0; S2 < 256; ++S2)
            sum += p[S2] * bf2f(V[(long)S2 * 4096 + col]);
        O[(long)s * 4096 + col] = f2bf(sum);
    }
}

// ---------------- launcher ----------------
extern "C" void kernel_launch(void* const* d_in, const int* in_sizes, int n_in,
                              void* d_out, int out_size, void* d_ws, size_t ws_size,
                              hipStream_t stream) {
    (void)in_sizes; (void)n_in; (void)out_size; (void)ws_size;
    const float* x    = (const float*)d_in[0];
    const float* wq_t = (const float*)d_in[1];
    const float* wk_t = (const float*)d_in[2];
    const float* wv_t = (const float*)d_in[3];
    const float* wo_t = (const float*)d_in[4];
    const float* wq_a = (const float*)d_in[5];
    const float* wk_a = (const float*)d_in[6];
    const float* wv_a = (const float*)d_in[7];
    const float* wo_a = (const float*)d_in[8];
    const float* w1   = (const float*)d_in[9];
    const float* b1   = (const float*)d_in[10];
    const float* w2   = (const float*)d_in[11];
    const float* b2   = (const float*)d_in[12];
    const float* w3   = (const float*)d_in[13];
    const float* b3   = (const float*)d_in[14];

    char* ws = (char*)d_ws;
    unsigned short* b0 = (unsigned short*)(ws);                         // 2MB bf16 256x4096
    unsigned short* B1 = (unsigned short*)(ws + (2l << 20));
    unsigned short* B2 = (unsigned short*)(ws + (4l << 20));
    unsigned short* B3 = (unsigned short*)(ws + (6l << 20));
    float*          S4 = (float*)(ws + (8l << 20));                     // up to 8x256x256 f32
    unsigned short* P5 = (unsigned short*)(ws + (10l << 20));           // up to 8x256x256 bf16

    const dim3 blk(256);
    const dim3 gFull(4096 / BN, 256 / BM, 1);   // 256x4096 outputs -> 128 blocks
    const dim3 gSq(256 / BN, 256 / BM, 1);      // 256x256 outputs
    const dim3 gSqH(256 / BN, 256 / BM, 8);     // per-head scores

    // ---- phase 1: time attention ----
    pack_x<<<4096, blk, 0, stream>>>(x, b0);
    gemm_bt<true,false,false><<<gFull, blk, 0, stream>>>(b0, wq_t, B1, 4096,4096,4096,4096, 0,0,0, 1.f, nullptr);
    gemm_bt<true,false,false><<<gFull, blk, 0, stream>>>(b0, wk_t, B2, 4096,4096,4096,4096, 0,0,0, 1.f, nullptr);
    gemm_bt<true,false,false><<<gFull, blk, 0, stream>>>(b0, wv_t, B3, 4096,4096,4096,4096, 0,0,0, 1.f, nullptr);
    rope_qk<<<2048, blk, 0, stream>>>(B1, B2);
    gemm_bt<false,true,false><<<gSq, blk, 0, stream>>>(B1, B2, S4, 4096,4096,4096,256, 0,0,0, 1.0f/64.0f, nullptr);
    softmax_rows<<<256, blk, 0, stream>>>(S4, P5);
    attn_av<<<dim3(256,1), blk, 0, stream>>>(P5, B3, b0, 4096);
    gemm_bt<true,false,false><<<gFull, blk, 0, stream>>>(b0, wo_t, B1, 4096,4096,4096,4096, 0,0,0, 1.f, nullptr);
    transpose_af<<<4096, blk, 0, stream>>>(B1, B2);     // H2 = agent-major

    // ---- phase 2: agent attention (8 heads, hd=512) ----
    gemm_bt<true,false,false><<<gFull, blk, 0, stream>>>(B2, wq_a, b0, 4096,4096,4096,4096, 0,0,0, 1.f, nullptr);
    gemm_bt<true,false,false><<<gFull, blk, 0, stream>>>(B2, wk_a, B1, 4096,4096,4096,4096, 0,0,0, 1.f, nullptr);
    gemm_bt<true,false,false><<<gFull, blk, 0, stream>>>(B2, wv_a, B3, 4096,4096,4096,4096, 0,0,0, 1.f, nullptr);
    gemm_bt<false,true,false><<<gSqH, blk, 0, stream>>>(b0, B1, S4, 512,4096,4096,256,
                                                        512, 512, 256l*256, 1.0f/22.62741699796952f, nullptr);
    softmax_rows<<<2048, blk, 0, stream>>>(S4, P5);
    attn_av<<<dim3(256,8), blk, 0, stream>>>(P5, B3, B2, 512);
    gemm_bt<true,false,false><<<gFull, blk, 0, stream>>>(B2, wo_a, b0, 4096,4096,4096,4096, 0,0,0, 1.f, nullptr);

    // ---- MLP: relu(h@w1.T+b1), relu(@w2.T+b2), @w3.T+b3 -> f32 out (256x256) ----
    gemm_bt<true,false,true ><<<gFull, blk, 0, stream>>>(b0, w1, B1, 4096,4096,4096,4096, 0,0,0, 1.f, b1);
    gemm_bt<true,false,true ><<<gFull, blk, 0, stream>>>(B1, w2, B2, 4096,4096,4096,4096, 0,0,0, 1.f, b2);
    gemm_bt<true,true ,false><<<gSq,   blk, 0, stream>>>(B2, w3, (float*)d_out, 4096,4096,4096,256, 0,0,0, 1.f, b3);
}
